// Set2SetPool_72816875536609
// MI455X (gfx1250) — matrix operand order repaired and checked by reading, not compiled, not run on hardware
//
#include <hip/hip_runtime.h>
#include <hip/hip_bf16.h>
#include <math.h>

// ---------------------------------------------------------------------------
// Set2Set pooling for MI455X (gfx1250, wave32).
//   B = out_size/256 graphs, H = 128, N = in_sizes[1] nodes, STEPS = 3.
// Per step:
//   1) gates = [q_star | h] (B x 384, bf16) @ Wcat (384 x 512, bf16 swizzled)
//      via v_wmma_f32_16x16x32_bf16, + bias, f32 accum.
//   2) LSTM pointwise -> c, h; h -> out[:,0:128] and bf16 A-matrix cols.
//   3) One-pass online segment softmax + weighted sum (wave32 per graph),
//      r -> out[:,128:256] and bf16 A-matrix cols 128..255.
// ---------------------------------------------------------------------------

typedef __bf16 bf16_t;
typedef __bf16 v16bf __attribute__((ext_vector_type(16)));
typedef __bf16 v8bf  __attribute__((ext_vector_type(8)));
typedef float  v8f   __attribute__((ext_vector_type(8)));

#define HDIM   128
#define FOURH  512
#define KDIM   384              // 2H (q_star) + H (h)
#define KTILES 12               // 384 / 32
#define NTILES 32               // 512 / 16
#define SWZW_ELEMS (NTILES * KTILES * 32 * 16)   // 196608 bf16

__device__ __forceinline__ float fsig(float x) { return 1.0f / (1.0f + __expf(-x)); }
__device__ __forceinline__ float ftanh(float x) {
  x = fminf(fmaxf(x, -15.0f), 15.0f);
  float e = __expf(2.0f * x);
  return (e - 1.0f) / (e + 1.0f);
}

// ---- setup kernels ---------------------------------------------------------

__global__ void zero_u32_kernel(unsigned int* __restrict__ p, size_t n) {
  size_t i = (size_t)blockIdx.x * blockDim.x + threadIdx.x;
  if (i < n) p[i] = 0u;
}

// Pack Wcat[k][n] (k<256: W_ih[n][k]; k>=256: W_hh[n][k-256]) into the exact
// per-lane layout of the WMMA B operand (32x16 bf16 tile):
//   element e of lane L in tile (nt, kt):  k = kt*32 + (e&7) + 8*(L/16) + 16*(e/8), n = nt*16 + (L&15)
//   flat index = ((nt*KTILES + kt)*32 + L)*16 + e   -> contiguous 32B per lane.
__global__ void pack_w_kernel(const float* __restrict__ Wih,
                              const float* __restrict__ Whh,
                              bf16_t* __restrict__ swzW) {
  int t = blockIdx.x * blockDim.x + threadIdx.x;
  if (t >= SWZW_ELEMS) return;
  int e    = t & 15;
  int lane = (t >> 4) & 31;
  int kt   = (t >> 9) % KTILES;
  int nt   = t / (KTILES * 512);
  int k = kt * 32 + (e & 7) + 8 * (lane >> 4) + 16 * (e >> 3);
  int n = nt * 16 + (lane & 15);
  float v = (k < 256) ? Wih[(size_t)n * 256 + k]
                      : Whh[(size_t)n * 128 + (k - 256)];
  swzW[t] = (bf16_t)v;
}

__global__ void bias_kernel(const float* __restrict__ a,
                            const float* __restrict__ b,
                            float* __restrict__ bias) {
  int i = blockIdx.x * blockDim.x + threadIdx.x;
  if (i < FOURH) bias[i] = a[i] + b[i];
}

// batch_vec is sorted: mark contiguous segment [start,end) per graph.
// ss/se pre-zeroed so empty graphs get start==end==0 (loop skipped, r = 0).
__global__ void seg_bounds_kernel(const int* __restrict__ bv,
                                  int* __restrict__ ss, int* __restrict__ se,
                                  int N) {
  int i = blockIdx.x * blockDim.x + threadIdx.x;
  if (i >= N) return;
  int b = bv[i];
  if (i == 0 || bv[i - 1] != b) ss[b] = i;
  if (i == N - 1 || bv[i + 1] != b) se[b] = i + 1;
}

// ---- step kernel 1: gates GEMM via WMMA -----------------------------------
// grid = ((B+15)/16, 2), block = 128 (4 waves). Each wave: 16(M) x 64(N) tile,
// K loop in steps of 32, 4 accumulators of v8f.
__global__ void gemm_gates_kernel(const bf16_t* __restrict__ A,
                                  const bf16_t* __restrict__ swzW,
                                  const float* __restrict__ bias,
                                  float* __restrict__ gates, int Brows) {
  const int lane = threadIdx.x & 31;
  const int wave = threadIdx.x >> 5;              // 0..3
  const int half = lane >> 4;
  const int l16  = lane & 15;
  const int m_base = blockIdx.x * 16;
  const int nt0 = blockIdx.y * 16 + wave * 4;     // base 16-wide N tile index

  int m = m_base + l16;
  if (m >= Brows) m = Brows - 1;                  // clamp loads; stores guarded
  const bf16_t* Arow = A + (size_t)m * KDIM;

  v8f acc0 = {}, acc1 = {}, acc2 = {}, acc3 = {};

#pragma unroll
  for (int kt = 0; kt < KTILES; ++kt) {
    const int kb = kt * 32;
    // 16-bit A 16x32 layout: lanes 0-15 hold k in {kb..kb+7, kb+16..kb+23},
    // lanes 16-31 hold k in {kb+8..kb+15, kb+24..kb+31}.
    v8bf alo = *(const v8bf*)(Arow + kb + 8 * half);
    v8bf ahi = *(const v8bf*)(Arow + kb + 16 + 8 * half);
    v16bf av;
#pragma unroll
    for (int e = 0; e < 8; ++e) { av[e] = alo[e]; av[e + 8] = ahi[e]; }

    const size_t wb = ((size_t)nt0 * KTILES + kt) * 512 + (size_t)lane * 16;
    v16bf b0 = *(const v16bf*)(swzW + wb);
    v16bf b1 = *(const v16bf*)(swzW + wb + (size_t)1 * KTILES * 512);
    v16bf b2 = *(const v16bf*)(swzW + wb + (size_t)2 * KTILES * 512);
    v16bf b3 = *(const v16bf*)(swzW + wb + (size_t)3 * KTILES * 512);

    acc0 = __builtin_amdgcn_wmma_f32_16x16x32_bf16(false, av, false, b0, (short)0, acc0, false, false);
    acc1 = __builtin_amdgcn_wmma_f32_16x16x32_bf16(false, av, false, b1, (short)0, acc1, false, false);
    acc2 = __builtin_amdgcn_wmma_f32_16x16x32_bf16(false, av, false, b2, (short)0, acc2, false, false);
    acc3 = __builtin_amdgcn_wmma_f32_16x16x32_bf16(false, av, false, b3, (short)0, acc3, false, false);
  }

  // C/D layout: lane L, component v -> M = v + 8*(L/16), N = (L&15).
  const float* bp = bias + nt0 * 16 + l16;
  const float bia0 = bp[0], bia1 = bp[16], bia2 = bp[32], bia3 = bp[48];
#pragma unroll
  for (int v = 0; v < 8; ++v) {
    int row = m_base + v + 8 * half;
    if (row < Brows) {
      float* g = gates + (size_t)row * FOURH + nt0 * 16 + l16;
      g[0]  = acc0[v] + bia0;
      g[16] = acc1[v] + bia1;
      g[32] = acc2[v] + bia2;
      g[48] = acc3[v] + bia3;
    }
  }
}

// ---- step kernel 2: LSTM pointwise ----------------------------------------
__global__ void lstm_kernel(const float* __restrict__ gates,
                            float* __restrict__ c, float* __restrict__ h,
                            bf16_t* __restrict__ Acat,
                            float* __restrict__ out, int Bn) {
  int idx = blockIdx.x * blockDim.x + threadIdx.x;
  if (idx >= Bn * HDIM) return;
  int b = idx >> 7;
  int j = idx & 127;
  const float* g = gates + (size_t)b * FOURH;
  float ig = fsig(g[j]);
  float fg = fsig(g[j + 128]);
  float gg = ftanh(g[j + 256]);
  float og = fsig(g[j + 384]);
  float cn = fg * c[idx] + ig * gg;
  float hn = og * ftanh(cn);
  c[idx] = cn;
  h[idx] = hn;
  out[(size_t)b * 256 + j] = hn;                 // q half of q_star
  bf16_t hb = (bf16_t)hn;
  Acat[(size_t)b * KDIM + j] = hb;               // q_star[:, 0:128] = h
  Acat[(size_t)b * KDIM + 256 + j] = hb;         // h operand of W_hh
}

// ---- step kernel 3: one-pass segment softmax + weighted sum ---------------
// One wave32 per graph; lane owns 4 channels (float4 -> 512B/row coalesced).
__global__ void attention_kernel(const float* __restrict__ x,
                                 const float* __restrict__ h,
                                 const int* __restrict__ seg_start,
                                 const int* __restrict__ seg_end,
                                 bf16_t* __restrict__ Acat,
                                 float* __restrict__ out, int Bn) {
  const int lane = threadIdx.x & 31;
  const int wave = threadIdx.x >> 5;
  const int g = blockIdx.x * 8 + wave;
  if (g >= Bn) return;
  const int s = seg_start[g];
  const int e = seg_end[g];
  const float4 qv = *(const float4*)(h + (size_t)g * HDIM + lane * 4);

  float m = -3.0e38f, denom = 0.0f;
  float ax = 0.0f, ay = 0.0f, az = 0.0f, aw = 0.0f;
  for (int node = s; node < e; ++node) {
    if (node + 8 < e)
      __builtin_prefetch(x + (size_t)(node + 8) * HDIM + lane * 4, 0, 0);
    float4 xv = *(const float4*)(x + (size_t)node * HDIM + lane * 4);
    float d = xv.x * qv.x + xv.y * qv.y + xv.z * qv.z + xv.w * qv.w;
    d += __shfl_xor(d, 16, 32);
    d += __shfl_xor(d, 8, 32);
    d += __shfl_xor(d, 4, 32);
    d += __shfl_xor(d, 2, 32);
    d += __shfl_xor(d, 1, 32);                    // e_i, replicated per lane
    float mn = fmaxf(m, d);
    float sc = __expf(m - mn);
    float p  = __expf(d - mn);
    denom = denom * sc + p;
    ax = ax * sc + p * xv.x;
    ay = ay * sc + p * xv.y;
    az = az * sc + p * xv.z;
    aw = aw * sc + p * xv.w;
    m = mn;
  }
  float inv = (denom > 0.0f) ? 1.0f / denom : 0.0f;
  float4 r;
  r.x = ax * inv; r.y = ay * inv; r.z = az * inv; r.w = aw * inv;
  *(float4*)(out + (size_t)g * 256 + 128 + lane * 4) = r;   // r half of q_star
  bf16_t* ar = Acat + (size_t)g * KDIM + 128 + lane * 4;
  ar[0] = (bf16_t)r.x; ar[1] = (bf16_t)r.y;
  ar[2] = (bf16_t)r.z; ar[3] = (bf16_t)r.w;
}

// ---------------------------------------------------------------------------

extern "C" void kernel_launch(void* const* d_in, const int* in_sizes, int n_in,
                              void* d_out, int out_size, void* d_ws, size_t ws_size,
                              hipStream_t stream) {
  (void)n_in; (void)ws_size;
  const float* x    = (const float*)d_in[0];
  const int*   bvec = (const int*)d_in[1];     // JAX x64 disabled -> int32
  const float* Wih  = (const float*)d_in[2];   // [512, 256]
  const float* Whh  = (const float*)d_in[3];   // [512, 128]
  const float* b_ih = (const float*)d_in[4];
  const float* b_hh = (const float*)d_in[5];
  const int N = in_sizes[1];
  const int B = out_size / 256;                // out = [B, 2H]
  float* out = (float*)d_out;

  // workspace carve-out (~63 MB)
  char* ws = (char*)d_ws;
  size_t off = 0;
  auto carve = [&](size_t bytes) -> void* {
    void* p = ws + off;
    off = (off + bytes + 255) & ~(size_t)255;
    return p;
  };
  bf16_t* swzW  = (bf16_t*)carve((size_t)SWZW_ELEMS * sizeof(bf16_t));
  float*  bias  = (float*) carve((size_t)FOURH * sizeof(float));
  int*    ss    = (int*)   carve((size_t)B * sizeof(int));
  int*    se    = (int*)   carve((size_t)B * sizeof(int));
  bf16_t* Acat  = (bf16_t*)carve((size_t)B * KDIM * sizeof(bf16_t));
  float*  gates = (float*) carve((size_t)B * FOURH * sizeof(float));
  float*  cst   = (float*) carve((size_t)B * HDIM * sizeof(float));
  float*  hst   = (float*) carve((size_t)B * HDIM * sizeof(float));

  // ---- setup (every call: harness does not re-initialize scratch) ----
  {
    size_t w;
    w = (size_t)B * KDIM * sizeof(bf16_t) / 4;   // zero Acat (q_star=h=0)
    zero_u32_kernel<<<(unsigned)((w + 255) / 256), 256, 0, stream>>>((unsigned int*)Acat, w);
    w = (size_t)B * HDIM;                         // zero c
    zero_u32_kernel<<<(unsigned)((w + 255) / 256), 256, 0, stream>>>((unsigned int*)cst, w);
    w = (size_t)B;                                // zero seg bounds
    zero_u32_kernel<<<(unsigned)((w + 255) / 256), 256, 0, stream>>>((unsigned int*)ss, w);
    zero_u32_kernel<<<(unsigned)((w + 255) / 256), 256, 0, stream>>>((unsigned int*)se, w);
  }
  pack_w_kernel<<<(SWZW_ELEMS + 255) / 256, 256, 0, stream>>>(Wih, Whh, swzW);
  bias_kernel<<<2, 256, 0, stream>>>(b_ih, b_hh, bias);
  seg_bounds_kernel<<<(N + 255) / 256, 256, 0, stream>>>(bvec, ss, se, N);

  // ---- 3 Set2Set steps ----
  dim3 ggrid((B + 15) / 16, 2);
  for (int step = 0; step < 3; ++step) {
    gemm_gates_kernel<<<ggrid, 128, 0, stream>>>(Acat, swzW, bias, gates, B);
    lstm_kernel<<<(B * HDIM + 255) / 256, 256, 0, stream>>>(gates, cst, hst, Acat, out, B);
    attention_kernel<<<(B + 7) / 8, 256, 0, stream>>>(x, hst, ss, se, Acat, out, B);
  }
}
